// StrongGNNMultiLabelOverlap_2310692405720
// MI455X (gfx1250) — compile-verified, matrix operand order checked by reading, not run on hardware
//
#include <hip/hip_runtime.h>
#include <hip/hip_bf16.h>
#include <math.h>

// ---------------- model constants (fixed by the reference) ----------------
#define HD    128        // hidden
#define NIN   64         // node feature in
#define EIN   16         // edge feature in
#define MGIN  2048       // morgan fingerprint dim
#define TD    1024       // target dim
#define NLB   1317       // labels
#define NNODE 65536
#define NEDGE 262144
#define NGR   2048
#define NPG   (NNODE / NGR)   // 32 nodes per graph (contiguous batch ids)
#define FEATK 519             // 4*HD+7
#define FEATS 544             // padded row stride (multiple of 32)

#define LN_EPS_ 1e-5f
#define BN_EPS_ 1e-5f
#define EPS_    1e-8f

#define BM 128           // rows per block (8 waves x 16)
#define BN 64            // cols per block
#define WTU (BN * 16)    // u32 words per W tile buffer

typedef __attribute__((ext_vector_type(16))) __bf16 v16bf;
typedef __attribute__((ext_vector_type(8)))  float  v8f;

// pack two f32 into two bf16 (truncation) in one v_perm_b32
__device__ __forceinline__ unsigned packbf2(float lo, float hi) {
  return __builtin_amdgcn_perm(__float_as_uint(hi), __float_as_uint(lo), 0x07060302u);
}

// pack 8 contiguous f32 -> 4 u32 (8 bf16); p must be 16B aligned
__device__ __forceinline__ void pack8(const float* __restrict__ p, unsigned* u) {
  const float4 x = *(const float4*)p;
  const float4 y = *(const float4*)(p + 4);
  u[0] = packbf2(x.x, x.y);
  u[1] = packbf2(x.z, x.w);
  u[2] = packbf2(y.x, y.y);
  u[3] = packbf2(y.z, y.w);
}

// A fragment (16x32 bf16) from contiguous f32 row chunk; elems 0..7 <- half*8+i,
// elems 8..15 <- 16+half*8+i  (ISA 7.12.2 16-bit A layout)
__device__ __forceinline__ v16bf packA(const float* __restrict__ p, int half) {
  union { v16bf v; unsigned u[8]; } au;
  pack8(p + half * 8,      au.u);
  pack8(p + 16 + half * 8, au.u + 4);
  return au.v;
}

// unguarded W chunk staging: wt[col*16+kk] = bf16{W[n][kb+2kk], W[n][kb+2kk+1]}
__device__ __forceinline__ void stageW(unsigned* wt, const float* __restrict__ W,
                                       int K, int nbase, int N, int kb, int tid) {
  #pragma unroll
  for (int u = tid; u < WTU; u += 256) {
    const int col = u >> 4;
    const int kk  = (u & 15) << 1;
    int n = nbase + col; if (n >= N) n = N - 1;
    const float* wr = W + (long)n * K + kb + kk;
    wt[u] = packbf2(wr[0], wr[1]);
  }
}

// load 4 B fragments from LDS, then 4 WMMAs
__device__ __forceinline__ void mma4(const v16bf a, const unsigned* wt,
                                     int rr, int half, v8f* acc) {
  uint4 q[8];
  #pragma unroll
  for (int t = 0; t < 4; ++t) {
    const int c = t * 16 + rr;
    q[2 * t]     = *(const uint4*)&wt[c * 16 + half * 4];
    q[2 * t + 1] = *(const uint4*)&wt[c * 16 + 8 + half * 4];
  }
  #pragma unroll
  for (int t = 0; t < 4; ++t) {
    union { v16bf v; uint4 q[2]; } bu;
    bu.q[0] = q[2 * t]; bu.q[1] = q[2 * t + 1];
    acc[t] = __builtin_amdgcn_wmma_f32_16x16x32_bf16(false, a, false, bu.v,
                                                     (short)0, acc[t], false, false);
  }
}

// --------------------------------------------------------------------------
// WMMA GEMM:  Out = act( [A1 | A2] * W^T + bias ),  W is [N, K] row-major.
// Block = 128 rows x 64 cols (8 waves, one N-tile). Double-buffered LDS W tile,
// one barrier per K-chunk. K-loop peeled into: full-A1 chunks, full-A2 chunks,
// and a single guarded tail chunk hoisted out of the loops.
// Call-site invariants: lda1/lda2 % 4 == 0; K1 % 32 == 0 whenever A2 != null;
// M % BM == 0.
// --------------------------------------------------------------------------
__global__ __launch_bounds__(256) void k_gemm(
    const float* __restrict__ A1, int K1, int lda1,
    const float* __restrict__ A2, int K2, int lda2,
    const int*   __restrict__ gidx,
    const float* __restrict__ W,
    const float* __restrict__ bias,
    float*       __restrict__ Out,
    const int*   __restrict__ sidx,
    int M, int N, int act)
{
  const int K = K1 + K2;
  const int nblocks = (N + BN - 1) / BN;
  const int bm = blockIdx.x / nblocks;
  const int bn = blockIdx.x - bm * nblocks;
  const int tid  = threadIdx.x;
  const int lane = tid & 31;
  const int half = lane >> 4;
  const int rr   = lane & 15;
  const int m0   = bm * BM + ((tid >> 5) << 4);
  const int nbase = bn * BN;

  __shared__ unsigned wtile[2][WTU];   // 2 x 4KB double-buffered W chunk

  int arow = m0 + rr; if (arow >= M) arow = M - 1;
  const int grow = gidx ? gidx[arow] : arow;
  const float* a1p = A1 + (long)grow * lda1;
  const float* a2p = A2 ? (A2 + (long)arow * lda2) : (const float*)0;

  v8f acc[4];
  #pragma unroll
  for (int t = 0; t < 4; ++t)
    #pragma unroll
    for (int i = 0; i < 8; ++i) acc[t][i] = 0.f;

  const int cFull = K >> 5;    // chunks fully inside [0, K)
  const int cA1   = K1 >> 5;   // chunks fully inside A1 (== K1/32; exact when A2 present)
  int ci = 0;

  // ---- phase 1: full chunks from A1 ----
  for (; ci < cA1; ++ci) {
    const int kb = ci << 5;
    unsigned* wt = wtile[ci & 1];
    stageW(wt, W, K, nbase, N, kb, tid);
    __syncthreads();
    mma4(packA(a1p + kb, half), wt, rr, half, acc);
  }
  // ---- phase 2: full chunks from A2 (empty when A2 == null) ----
  for (; ci < cFull; ++ci) {
    const int kb = ci << 5;
    unsigned* wt = wtile[ci & 1];
    stageW(wt, W, K, nbase, N, kb, tid);
    __syncthreads();
    mma4(packA(a2p + (kb - K1), half), wt, rr, half, acc);
  }
  // ---- phase 3: single guarded tail chunk ----
  if ((cFull << 5) < K) {
    const int kb = cFull << 5;
    unsigned* wt = wtile[cFull & 1];
    #pragma unroll
    for (int u = tid; u < WTU; u += 256) {
      const int col = u >> 4;
      const int kk  = (u & 15) << 1;
      int n = nbase + col; if (n >= N) n = N - 1;
      const float* wr = W + (long)n * K + kb + kk;
      const float lo = (kb + kk     < K) ? wr[0] : 0.f;
      const float hi = (kb + kk + 1 < K) ? wr[1] : 0.f;
      wt[u] = packbf2(lo, hi);
    }
    __syncthreads();
    union { v16bf v; unsigned u[8]; } au;
    #pragma unroll
    for (int j = 0; j < 8; ++j) {
      const int k0 = kb + ((j < 4) ? (half * 8 + 2 * j) : (16 + half * 8 + 2 * (j - 4)));
      float lo = 0.f, hi = 0.f;
      if      (k0 < K1)     lo = a1p[k0];
      else if (k0 < K)      lo = a2p[k0 - K1];
      if      (k0 + 1 < K1) hi = a1p[k0 + 1];
      else if (k0 + 1 < K)  hi = a2p[k0 + 1 - K1];
      au.u[j] = packbf2(lo, hi);
    }
    mma4(au.v, wt, rr, half, acc);
  }

  // ---- epilogue: lane holds column (nbase+t*16+rr), rows m0 + half*8 + v
  #pragma unroll
  for (int t = 0; t < 4; ++t) {
    const int n = nbase + t * 16 + rr;
    if (n >= N) continue;
    const float bv = bias ? bias[n] : 0.f;
    #pragma unroll
    for (int v = 0; v < 8; ++v) {
      const int m = m0 + half * 8 + v;
      if (m >= M) continue;
      float val = acc[t][v] + bv;
      if (act == 1)      val = fmaxf(val, 0.f);
      else if (act == 2) val = 1.f / (1.f + __expf(-val));
      if (sidx) unsafeAtomicAdd(&Out[(long)sidx[m] * N + n], val);
      else      Out[(long)m * N + n] = val;
    }
  }
}

__global__ void k_fill0(float* __restrict__ p, long n) {
  long i = (long)blockIdx.x * blockDim.x + threadIdx.x;
  if (i < n) p[i] = 0.f;
}

// GRUCell (gate order r,z,n) + eval BatchNorm + residual, in-place on h
__global__ void k_gru_bn(const float* __restrict__ gi, const float* __restrict__ gh,
                         float* __restrict__ h,
                         const float* __restrict__ bg, const float* __restrict__ bb,
                         const float* __restrict__ brm, const float* __restrict__ brv,
                         int Mrows)
{
  long idx = (long)blockIdx.x * blockDim.x + threadIdx.x;
  if (idx >= (long)Mrows * HD) return;
  const int nrow = (int)(idx >> 7);
  const int j    = (int)(idx & (HD - 1));
  const float* gip = gi + (long)nrow * 3 * HD;
  const float* ghp = gh + (long)nrow * 3 * HD;
  const float r  = 1.f / (1.f + __expf(-(gip[j]      + ghp[j])));
  const float z  = 1.f / (1.f + __expf(-(gip[HD + j] + ghp[HD + j])));
  const float nn = tanhf(gip[2 * HD + j] + r * ghp[2 * HD + j]);
  const float hold = h[idx];
  const float hnew = (1.f - z) * nn + z * hold;
  const float bn = (hnew - brm[j]) * rsqrtf(brv[j] + BN_EPS_) * bg[j] + bb[j];
  h[idx] = hold + bn;
}

// segment mean/max readout; batch ids are contiguous (NPG nodes per graph)
__global__ void k_readout(const float* __restrict__ h, float* __restrict__ zcat, int npg)
{
  const int g = blockIdx.x, j = threadIdx.x;          // blockDim = HD
  const float* base = h + (long)g * npg * HD;
  float s = 0.f, mx = -3.402823e38f;
  for (int i = 0; i < npg; ++i) {
    const float v = base[(long)i * HD + j];
    s += v; mx = fmaxf(mx, v);
  }
  zcat[(long)g * (2 * HD) + j]      = s / (float)npg;
  zcat[(long)g * (2 * HD) + HD + j] = mx;
}

// mode 0: x = g*a + (1-g)*b (GatedFusion)   mode 1: x = a + g*b (CrossDrugGate)
// then LayerNorm(x). One 128-thread block per row.
__global__ void k_fuse_ln(const float* __restrict__ a, const float* __restrict__ b,
                          const float* __restrict__ g,
                          const float* __restrict__ gam, const float* __restrict__ bet,
                          float* __restrict__ out, int mode)
{
  const int row = blockIdx.x;
  const int j   = threadIdx.x;                        // HD threads
  const long base = (long)row * HD;
  const float gv = g[base + j], av = a[base + j], bv = b[base + j];
  const float x = (mode == 0) ? (gv * av + (1.f - gv) * bv) : (av + gv * bv);
  __shared__ float s1[HD], s2[HD];
  s1[j] = x; s2[j] = x * x;
  __syncthreads();
  for (int off = HD / 2; off > 0; off >>= 1) {
    if (j < off) { s1[j] += s1[j + off]; s2[j] += s2[j + off]; }
    __syncthreads();
  }
  const float mu  = s1[0] * (1.f / HD);
  const float var = s2[0] * (1.f / HD) - mu * mu;
  out[base + j] = (x - mu) * rsqrtf(var + LN_EPS_) * gam[j] + bet[j];
}

__device__ __forceinline__ float wred(float v) {
  for (int o = 16; o > 0; o >>= 1) v += __shfl_down(v, o, 32);
  return v;
}

// 7 overlap features per graph-pair row; one wave per row
__global__ void k_overlap(const float* __restrict__ t1, const float* __restrict__ t2,
                          float* __restrict__ ov, int D)
{
  const int row  = blockIdx.x * (blockDim.x >> 5) + (threadIdx.x >> 5);
  const int lane = threadIdx.x & 31;
  if (row >= NGR) return;
  const float* a = t1 + (long)row * D;
  const float* b = t2 + (long)row * D;
  float sh = 0.f, c1 = 0.f, c2 = 0.f, un = 0.f, n1 = 0.f, n2 = 0.f;
  for (int i = lane; i < D; i += 32) {
    const float x = a[i], y = b[i];
    sh += x * y; c1 += x; c2 += y;
    un += ((x + y) > 0.f) ? 1.f : 0.f;
    n1 += x * x; n2 += y * y;
  }
  sh = wred(sh); c1 = wred(c1); c2 = wred(c2);
  un = wred(un); n1 = wred(n1); n2 = wred(n2);
  if (lane == 0) {
    float* o7 = ov + (long)row * 7;
    o7[0] = log1pf(sh);
    o7[1] = log1pf(un);
    o7[2] = sh / (un + EPS_);
    o7[3] = sh / (sqrtf(n1 + EPS_) * sqrtf(n2 + EPS_) + EPS_);
    o7[4] = (sh > 0.f) ? 1.f : 0.f;
    o7[5] = log1pf(c1);
    o7[6] = log1pf(c2);
  }
}

// feat = [z1n | z2n | |z1n-z2n| | z1n*z2n | ov | 0-pad] -> [NGR, FEATS]
__global__ void k_feat(const float* __restrict__ z1, const float* __restrict__ z2,
                       const float* __restrict__ ov, float* __restrict__ feat)
{
  long i = (long)blockIdx.x * blockDim.x + threadIdx.x;
  if (i >= (long)NGR * FEATS) return;
  const int row = (int)(i / FEATS);
  const int c   = (int)(i - (long)row * FEATS);
  const long b  = (long)row * HD;
  float v;
  if      (c < HD)      v = z1[b + c];
  else if (c < 2 * HD)  v = z2[b + c - HD];
  else if (c < 3 * HD)  v = fabsf(z1[b + c - 2 * HD] - z2[b + c - 2 * HD]);
  else if (c < 4 * HD)  v = z1[b + c - 3 * HD] * z2[b + c - 3 * HD];
  else if (c < FEATK)   v = ov[(long)row * 7 + (c - 4 * HD)];
  else                  v = 0.f;   // stride padding
  feat[i] = v;
}

// --------------------------------------------------------------------------
extern "C" void kernel_launch(void* const* d_in, const int* in_sizes, int n_in,
                              void* d_out, int out_size, void* d_ws, size_t ws_size,
                              hipStream_t stream)
{
  (void)in_sizes; (void)n_in; (void)out_size; (void)ws_size;
  auto F = [&](int i) -> const float* { return (const float*)d_in[i]; };

  // params in setup_inputs() dict insertion order
  const float *lin_inW = F(0), *lin_inb = F(1);
  struct Lay { const float *m1W,*m1b,*m2W,*m2b,*bg,*bb,*brm,*brv,*Wih,*Whh,*bih,*bhh; };
  Lay L[4];
  for (int i = 0; i < 4; ++i) {
    const int b = 2 + 12 * i;
    L[i] = { F(b+0),F(b+1),F(b+2),F(b+3),F(b+4),F(b+5),F(b+6),F(b+7),
             F(b+8),F(b+9),F(b+10),F(b+11) };
  }
  const float *roW=F(50), *rob=F(51);
  const float *mg1W=F(52), *mg1b=F(53), *mg2W=F(54), *mg2b=F(55);
  const float *fu1W=F(56), *fu1b=F(57), *fu2W=F(58), *fu2b=F(59), *fu_g=F(60), *fu_b=F(61);
  const float *g12_1W=F(62), *g12_1b=F(63), *g12_2W=F(64), *g12_2b=F(65);
  const float *g21_1W=F(66), *g21_1b=F(67), *g21_2W=F(68), *g21_2b=F(69);
  const float *ln1g=F(70), *ln1b=F(71), *ln2g=F(72), *ln2b=F(73);
  const float *hd1W=F(74), *hd1b=F(75), *hd2W=F(76), *hd2b=F(77), *hd3W=F(78), *hd3b=F(79);
  const float *x1 = F(80);  const int *ei1 = (const int*)d_in[81]; const float *ea1 = F(82);
  const float *x2 = F(84);  const int *ei2 = (const int*)d_in[85]; const float *ea2 = F(86);
  const float *mo1 = F(88), *mo2 = F(89), *tg1 = F(90), *tg2 = F(91);

  // ---- workspace carve (floats) ----
  float* ws = (float*)d_ws;
  size_t off = 0;
  auto alloc = [&](size_t n) { float* p = ws + off; off += n; return p; };
  float* bufE = alloc((size_t)NEDGE * HD);       // msg1 output; aliased as gi later
  float* gh   = alloc((size_t)NNODE * 3 * HD);
  float* h    = alloc((size_t)NNODE * HD);
  float* agg  = alloc((size_t)NNODE * HD);
  float* zcat = alloc((size_t)NGR * 2 * HD);
  float* zg[2] = { alloc((size_t)NGR * HD), alloc((size_t)NGR * HD) };
  float* za[2] = { alloc((size_t)NGR * HD), alloc((size_t)NGR * HD) };
  float* zf[2] = { alloc((size_t)NGR * HD), alloc((size_t)NGR * HD) };
  float* zn[2] = { alloc((size_t)NGR * HD), alloc((size_t)NGR * HD) };
  float* gA   = alloc((size_t)NGR * HD);
  float* tA   = alloc((size_t)NGR * HD);
  float* tB   = alloc((size_t)NGR * 2 * HD);
  float* ov   = alloc((size_t)NGR * 7);
  float* feat = alloc((size_t)NGR * FEATS);
  float* gi   = bufE;                            // NNODE*3*HD <= NEDGE*HD

  auto gemm = [&](const float* A1, int K1, int lda1,
                  const float* A2, int K2, int lda2, const int* gx,
                  const float* Wm, const float* bv, float* Out, const int* sx,
                  int M, int N, int act) {
    const int blocks = ((M + BM - 1) / BM) * ((N + BN - 1) / BN);
    hipLaunchKernelGGL(k_gemm, dim3(blocks), dim3(256), 0, stream,
                       A1, K1, lda1, A2, K2, lda2, gx, Wm, bv, Out, sx, M, N, act);
  };
  auto fill0 = [&](float* p, long n) {
    hipLaunchKernelGGL(k_fill0, dim3((int)((n + 255) / 256)), dim3(256), 0, stream, p, n);
  };

  auto encode = [&](const float* x, const int* ei, const float* ea, float* zout) {
    gemm(x, NIN, NIN, nullptr, 0, 0, nullptr, lin_inW, lin_inb, h, nullptr, NNODE, HD, 1);
    const int* src = ei;
    const int* dst = ei + NEDGE;
    for (int l = 0; l < 4; ++l) {
      // msg1: relu(lin([h[src] | edge_attr]))   (gathered concat A)
      gemm(h, HD, HD, ea, EIN, EIN, src, L[l].m1W, L[l].m1b, bufE, nullptr, NEDGE, HD, 1);
      // msg2 fused with segment_sum scatter into agg
      fill0(agg, (long)NNODE * HD);
      gemm(bufE, HD, HD, nullptr, 0, 0, nullptr, L[l].m2W, L[l].m2b, agg, dst, NEDGE, HD, 0);
      // GRU gate GEMMs
      gemm(agg, HD, HD, nullptr, 0, 0, nullptr, L[l].Wih, L[l].bih, gi, nullptr, NNODE, 3 * HD, 0);
      gemm(h,   HD, HD, nullptr, 0, 0, nullptr, L[l].Whh, L[l].bhh, gh, nullptr, NNODE, 3 * HD, 0);
      hipLaunchKernelGGL(k_gru_bn, dim3((NNODE * HD + 255) / 256), dim3(256), 0, stream,
                         gi, gh, h, L[l].bg, L[l].bb, L[l].brm, L[l].brv, NNODE);
    }
    hipLaunchKernelGGL(k_readout, dim3(NGR), dim3(HD), 0, stream, h, zcat, NPG);
    gemm(zcat, 2 * HD, 2 * HD, nullptr, 0, 0, nullptr, roW, rob, zout, nullptr, NGR, HD, 1);
  };

  encode(x1, ei1, ea1, zg[0]);
  encode(x2, ei2, ea2, zg[1]);

  // AuxEncoder (morgan)
  for (int d = 0; d < 2; ++d) {
    const float* mo = d ? mo2 : mo1;
    gemm(mo, MGIN, MGIN, nullptr, 0, 0, nullptr, mg1W, mg1b, tA, nullptr, NGR, HD, 1);
    gemm(tA, HD, HD, nullptr, 0, 0, nullptr, mg2W, mg2b, za[d], nullptr, NGR, HD, 1);
  }
  // GatedFusion
  for (int d = 0; d < 2; ++d) {
    gemm(zg[d], HD, HD, za[d], HD, HD, nullptr, fu1W, fu1b, tA, nullptr, NGR, HD, 1);
    gemm(tA, HD, HD, nullptr, 0, 0, nullptr, fu2W, fu2b, gA, nullptr, NGR, HD, 2);
    hipLaunchKernelGGL(k_fuse_ln, dim3(NGR), dim3(HD), 0, stream,
                       zg[d], za[d], gA, fu_g, fu_b, zf[d], 0);
  }
  // CrossDrugGate
  gemm(zf[0], HD, HD, zf[1], HD, HD, nullptr, g12_1W, g12_1b, tA, nullptr, NGR, HD, 1);
  gemm(tA, HD, HD, nullptr, 0, 0, nullptr, g12_2W, g12_2b, gA, nullptr, NGR, HD, 2);
  hipLaunchKernelGGL(k_fuse_ln, dim3(NGR), dim3(HD), 0, stream,
                     zf[0], zf[1], gA, ln1g, ln1b, zn[0], 1);
  gemm(zf[1], HD, HD, zf[0], HD, HD, nullptr, g21_1W, g21_1b, tA, nullptr, NGR, HD, 1);
  gemm(tA, HD, HD, nullptr, 0, 0, nullptr, g21_2W, g21_2b, gA, nullptr, NGR, HD, 2);
  hipLaunchKernelGGL(k_fuse_ln, dim3(NGR), dim3(HD), 0, stream,
                     zf[1], zf[0], gA, ln2g, ln2b, zn[1], 1);

  // PairHead
  hipLaunchKernelGGL(k_overlap, dim3((NGR + 7) / 8), dim3(256), 0, stream, tg1, tg2, ov, TD);
  {
    const long n = (long)NGR * FEATS;
    hipLaunchKernelGGL(k_feat, dim3((int)((n + 255) / 256)), dim3(256), 0, stream,
                       zn[0], zn[1], ov, feat);
  }
  gemm(feat, FEATK, FEATS, nullptr, 0, 0, nullptr, hd1W, hd1b, tB, nullptr, NGR, 2 * HD, 1);
  gemm(tB, 2 * HD, 2 * HD, nullptr, 0, 0, nullptr, hd2W, hd2b, tA, nullptr, NGR, HD, 1);
  gemm(tA, HD, HD, nullptr, 0, 0, nullptr, hd3W, hd3b, (float*)d_out, nullptr, NGR, NLB, 0);
}
